// Conv_attn_16131897164206
// MI455X (gfx1250) — matrix operand order repaired and checked by reading, not compiled, not run on hardware
//
#include <hip/hip_runtime.h>
#include <cstdint>

// MI455X / gfx1250 fused Swin-window-attention.
//   Kernel 1: split fp32 weights into f16 hi/lo (one-time per launch, tiny).
//   Kernel 2: per (shift, window) fused  QKV-GEMM -> RoPE -> attention -> y
//             (y stored as packed {hi,lo} f16 pairs in u32, fp32-class precision).
//   Kernel 3: proj GEMM  y @ proj_w^T + b -> out, A-tiles staged to LDS with
//             GLOBAL_LOAD_ASYNC_TO_LDS_B128 (ASYNCcnt) — CDNA5 async path.
// All GEMMs use v_wmma_f32_16x16x32_f16 with the f16 "hi+lo" 3-product split
// (error ~2^-22) — 8x more K per WMMA than the native f32 WMMA.
// Workspace layout (bytes):
//   [0, 134217728)   y packed hi/lo u32 : 8*64*64*1024
//   [+ ..)           qkv_w hi/lo (f16), proj_w hi/lo (f16)  (~17 MB)

typedef __attribute__((ext_vector_type(16))) _Float16 v16h;
typedef __attribute__((ext_vector_type(8)))  float    v8f;

#define WMMA_F16(a, b, c) \
  __builtin_amdgcn_wmma_f32_16x16x32_f16(false, (a), false, (b), (short)0, (c), false, false)

__device__ __forceinline__ v8f wmma3(v16h ah, v16h al, v16h bh, v16h bl, v8f c) {
  // (ah+al)*(bh+bl) ~= al*bh + ah*bl + ah*bh   (drop al*bl ~ 2^-22)
  c = WMMA_F16(al, bh, c);
  c = WMMA_F16(ah, bl, c);
  c = WMMA_F16(ah, bh, c);
  return c;
}

struct HL { _Float16 h, l; };
__device__ __forceinline__ HL split2(float x) {
  const _Float16 hh = (_Float16)x;
  HL r;
  r.h = hh;
  r.l = (_Float16)(x - (float)hh);
  return r;
}
__device__ __forceinline__ unsigned pack_hl(float x) {
  const HL a = split2(x);
  return (unsigned)__builtin_bit_cast(unsigned short, a.h) |
         ((unsigned)__builtin_bit_cast(unsigned short, a.l) << 16);
}
__device__ __forceinline__ _Float16 f16_lo(unsigned u) {
  return __builtin_bit_cast(_Float16, (unsigned short)(u & 0xffffu));
}
__device__ __forceinline__ _Float16 f16_hi(unsigned u) {
  return __builtin_bit_cast(_Float16, (unsigned short)(u >> 16));
}

// ---- CDNA5 async global->LDS copy (16B per lane), ASYNCcnt-tracked ---------
__device__ __forceinline__ void async_copy_b128(unsigned lds_off, const void* src) {
  asm volatile("global_load_async_to_lds_b128 %0, %1, off"
               :: "v"(lds_off), "v"((unsigned long long)(uintptr_t)src)
               : "memory");
}
__device__ __forceinline__ void wait_async0() {
  asm volatile("s_wait_asynccnt 0" ::: "memory");
}

// ---- WMMA fragment loaders (ISA 7.12.2 layouts, wave32) --------------------
// A 16x32 f16: lanes 0-15 -> M=lane, K in {0..7,16..23}; lanes 16-31 -> +8 K off.
__device__ __forceinline__ void load_a_h(const _Float16* hi, const _Float16* lo, int ld,
                                         int row0, int k0, int lane, v16h &ah, v16h &al) {
  const int hf = (lane >> 4) & 1, lr = lane & 15;
  const _Float16* ph = hi + (size_t)(row0 + lr) * ld + k0 + 8 * hf;
  const _Float16* pl = lo + (size_t)(row0 + lr) * ld + k0 + 8 * hf;
#pragma unroll
  for (int j = 0; j < 8; ++j) {
    const int kk = (j < 4) ? 2 * j : 16 + 2 * (j - 4);
    ah[2*j] = ph[kk]; ah[2*j+1] = ph[kk+1];
    al[2*j] = pl[kk]; al[2*j+1] = pl[kk+1];
  }
}

// A fragment from packed {hi,lo} u32 source (e.g. y tile in LDS)
__device__ __forceinline__ void load_a_pk(const unsigned* src, int ld, int row0, int k0,
                                          int lane, v16h &ah, v16h &al) {
  const int hf = (lane >> 4) & 1, lr = lane & 15;
  const unsigned* p = src + (size_t)(row0 + lr) * ld + k0 + 8 * hf;
#pragma unroll
  for (int j = 0; j < 8; ++j) {
    const int kk = (j < 4) ? 2 * j : 16 + 2 * (j - 4);
    const unsigned u0 = p[kk], u1 = p[kk + 1];
    ah[2*j]   = f16_lo(u0); al[2*j]   = f16_hi(u0);
    ah[2*j+1] = f16_lo(u1); al[2*j+1] = f16_hi(u1);
  }
}

__device__ __forceinline__ void load_a_f32(const float* src, int ld, int row0, int k0,
                                           int lane, v16h &ah, v16h &al) {
  const int hf = (lane >> 4) & 1, lr = lane & 15;
  const float* p = src + (size_t)(row0 + lr) * ld + k0 + 8 * hf;
#pragma unroll
  for (int j = 0; j < 8; ++j) {
    const int kk = (j < 4) ? 2 * j : 16 + 2 * (j - 4);
    const HL a = split2(p[kk]);
    const HL b = split2(p[kk + 1]);
    ah[2*j]   = a.h; al[2*j]   = a.l;
    ah[2*j+1] = b.h; al[2*j+1] = b.l;
  }
}

// B 32x16 f16: lanes 0-15 -> N=lane, K in {0..7,16..23}; lanes 16-31 -> +8 K off.
// B[k][n] = W[n][k]  (weight row-major over k)
__device__ __forceinline__ void load_b_wT_h(const _Float16* hi, const _Float16* lo, int ld,
                                            int k0, int n0, int lane, v16h &bh, v16h &bl) {
  const int hf = (lane >> 4) & 1, lr = lane & 15;
  const _Float16* ph = hi + (size_t)(n0 + lr) * ld + k0 + 8 * hf;
  const _Float16* pl = lo + (size_t)(n0 + lr) * ld + k0 + 8 * hf;
#pragma unroll
  for (int j = 0; j < 16; ++j) { const int kk = (j < 8) ? j : j + 8; bh[j] = ph[kk]; bl[j] = pl[kk]; }
}

// B[k][n] = src[n][k]  (f32, split on the fly) e.g. K^T
__device__ __forceinline__ void load_b_T_f32(const float* src, int ld, int k0, int n0,
                                             int lane, v16h &bh, v16h &bl) {
  const int hf = (lane >> 4) & 1, lr = lane & 15;
  const float* p = src + (size_t)(n0 + lr) * ld + k0 + 8 * hf;
#pragma unroll
  for (int j = 0; j < 16; ++j) {
    const HL a = split2(p[(j < 8) ? j : j + 8]);
    bh[j] = a.h; bl[j] = a.l;
  }
}

// B[k][n] = src[k][n]  (f32, split on the fly) e.g. V
__device__ __forceinline__ void load_b_f32(const float* src, int ld, int k0, int n0,
                                           int lane, v16h &bh, v16h &bl) {
  const int hf = (lane >> 4) & 1, lr = lane & 15;
  const float* p = src + (size_t)(k0 + 8 * hf) * ld + n0 + lr;
#pragma unroll
  for (int j = 0; j < 16; ++j) {
    const HL a = split2(p[(size_t)((j < 8) ? j : j + 8) * ld]);
    bh[j] = a.h; bl[j] = a.l;
  }
}

// Swin shift-mask region id along one axis (see make_mask in reference)
__device__ __forceinline__ int region1d(int c, int s) {
  if (s == 0) return 0;
  if (c < 56) return 0;            // H - ws
  return (c < 64 - s) ? 1 : 2;     // [H-ws, H-s) vs [H-s, H)
}

// ---- Kernel 1: split fp32 weight matrix into f16 hi/lo ---------------------
__global__ __launch_bounds__(256) void wsplit(const float* __restrict__ w,
                                              _Float16* __restrict__ hi,
                                              _Float16* __restrict__ lo, int n) {
  int i = blockIdx.x * 256 + threadIdx.x;
  if (i < n) {
    const HL a = split2(w[i]);
    hi[i] = a.h;
    lo[i] = a.l;
  }
}

// ---- Kernel 2: fused QKV-GEMM + RoPE + window attention --------------------
// grid = (512 windows, 4 shifts), block = 256 (8 waves)
#define XLD 1032  // padded row stride (f16) -> 4-bank stagger, conflict-free A loads
__global__ __launch_bounds__(256) void attn_fused(
    const float* __restrict__ x,
    const _Float16* __restrict__ qwhi, const _Float16* __restrict__ qwlo,
    const float* __restrict__ qkv_b,
    unsigned* __restrict__ y) {
  __shared__ union {
    struct { _Float16 xhi[64][XLD]; _Float16 xlo[64][XLD]; } g;  // 258 KB
    struct { float qkv[64][768]; float S[8][64][64]; } a;        // 320 KB
  } sm;

  const int tid = threadIdx.x;
  const int lane = tid & 31, wave = tid >> 5;
  const int hf = (lane >> 4) & 1, lr = lane & 15;
  const int shift = blockIdx.y;
  const int s0 = (shift & 2) ? 4 : 0;
  const int s1 = (shift & 1) ? 4 : 0;
  const int win = blockIdx.x;
  const int b = win >> 6, wy = (win >> 3) & 7, wx = win & 7;

  // Stage shifted x window into LDS as f16 hi/lo (coalesced along channels).
  for (int i = tid; i < 64 * 1024; i += 256) {
    const int row = i >> 10, k = i & 1023;
    const int hs = wy * 8 + (row >> 3), wsS = wx * 8 + (row & 7);
    const int hsrc = (hs + s0) & 63, wsrc = (wsS + s1) & 63;
    const float v = x[(((size_t)b * 64 + hsrc) * 64 + wsrc) * 1024 + k];
    const HL a = split2(v);
    sm.g.xhi[row][k] = a.h;
    sm.g.xlo[row][k] = a.l;
  }
  __syncthreads();

  // QKV GEMM: 64 x 768 (this shift's weight slice), K = 1024.
  const int nb = wave * 96;  // each wave owns 96 output columns (6 N-tiles)
  const _Float16* Wh = qwhi + (size_t)(shift * 768) * 1024;
  const _Float16* Wl = qwlo + (size_t)(shift * 768) * 1024;

  v8f acc[4][6] = {};
#pragma unroll 1
  for (int k0 = 0; k0 < 1024; k0 += 32) {
    v16h ah[4], al[4];
#pragma unroll
    for (int mi = 0; mi < 4; ++mi)
      load_a_h(&sm.g.xhi[0][0], &sm.g.xlo[0][0], XLD, mi * 16, k0, lane, ah[mi], al[mi]);
#pragma unroll
    for (int nt = 0; nt < 6; ++nt) {
      v16h bh, bl;
      load_b_wT_h(Wh, Wl, 1024, k0, nb + nt * 16, lane, bh, bl);
#pragma unroll
      for (int mi = 0; mi < 4; ++mi)
        acc[mi][nt] = wmma3(ah[mi], al[mi], bh, bl, acc[mi][nt]);
    }
  }
  __syncthreads();  // all waves finished reading x before LDS union reuse

  // Bias, park qkv (fp32) in LDS.  C/D layout: VGPR v -> M = v + 8*hf, N = lr.
#pragma unroll
  for (int nt = 0; nt < 6; ++nt) {
    const float bias = qkv_b[shift * 768 + nb + nt * 16 + lr];
#pragma unroll
    for (int mi = 0; mi < 4; ++mi)
#pragma unroll
      for (int v = 0; v < 8; ++v)
        sm.a.qkv[mi * 16 + v + 8 * hf][nb + nt * 16 + lr] = acc[mi][nt][v] + bias;
  }
  __syncthreads();

  // RoPE on q,k (+ q scale).  hd=32: dims [0,16) rotate by ty, [16,32) by tx.
  const float scale = 0.17677669529663687f;  // 32^-0.5
  for (int t = tid; t < 64 * 8 * 16; t += 256) {
    const int pair = t & 15, ht = t >> 4;
    const int head = ht & 7, tok = ht >> 3;
    const int axis = pair >> 3, j = pair & 7;
    const int ty = tok >> 3, tx = tok & 7;
    const float pos = (float)(axis ? tx : ty);
    const float ang = pos * __expf(-1.1512925465f * (float)j);  // 10000^(-j/8)
    float sn, cs;
    __sincosf(ang, &sn, &cs);
    const int d0 = axis * 16 + 2 * j;
    float* qr = &sm.a.qkv[tok][head * 32];
    float t1 = qr[d0], t2 = qr[d0 + 1];
    qr[d0]     = (t1 * cs - t2 * sn) * scale;
    qr[d0 + 1] = (t1 * sn + t2 * cs) * scale;
    float* kr = &sm.a.qkv[tok][256 + head * 32];
    t1 = kr[d0]; t2 = kr[d0 + 1];
    kr[d0]     = t1 * cs - t2 * sn;
    kr[d0 + 1] = t1 * sn + t2 * cs;
  }
  __syncthreads();

  // Attention: one head per wave.  S = q k^T  (64x64, K=32 -> one f16x3 step).
  const int h = wave;
  v8f sacc[4][4] = {};
  {
    v16h qah[4], qal[4];
#pragma unroll
    for (int mi = 0; mi < 4; ++mi)
      load_a_f32(&sm.a.qkv[0][h * 32], 768, mi * 16, 0, lane, qah[mi], qal[mi]);
#pragma unroll
    for (int ni = 0; ni < 4; ++ni) {
      v16h bh, bl;
      load_b_T_f32(&sm.a.qkv[0][256 + h * 32], 768, 0, ni * 16, lane, bh, bl);
#pragma unroll
      for (int mi = 0; mi < 4; ++mi)
        sacc[mi][ni] = wmma3(qah[mi], qal[mi], bh, bl, sacc[mi][ni]);
    }
  }
  // Shift mask (computed analytically) + park scores in LDS.
#pragma unroll
  for (int mi = 0; mi < 4; ++mi)
#pragma unroll
    for (int ni = 0; ni < 4; ++ni)
#pragma unroll
      for (int v = 0; v < 8; ++v) {
        const int row = mi * 16 + v + 8 * hf, col = ni * 16 + lr;
        float sv = sacc[mi][ni][v];
        if (shift) {
          const int rr = region1d(wy * 8 + (row >> 3), s0) * 3 +
                         region1d(wx * 8 + (row & 7), s1);
          const int rc = region1d(wy * 8 + (col >> 3), s0) * 3 +
                         region1d(wx * 8 + (col & 7), s1);
          if (rr != rc) sv -= 100.0f;
        }
        sm.a.S[h][row][col] = sv;
      }

  // Softmax over rows of this wave's own head (LDS ops are in-order per wave).
  for (int r = lane; r < 64; r += 32) {
    float mx = -3.0e38f;
    for (int c = 0; c < 64; ++c) mx = fmaxf(mx, sm.a.S[h][r][c]);
    float sum = 0.f;
    for (int c = 0; c < 64; ++c) {
      const float e = __expf(sm.a.S[h][r][c] - mx);
      sm.a.S[h][r][c] = e;
      sum += e;
    }
    const float inv = 1.0f / sum;
    for (int c = 0; c < 64; ++c) sm.a.S[h][r][c] *= inv;
  }

  // out = P V   (64x32, K=64 -> two f16x3 steps)
  v8f oacc[4][2] = {};
#pragma unroll
  for (int ks = 0; ks < 64; ks += 32) {
    v16h pah[4], pal[4];
#pragma unroll
    for (int mi = 0; mi < 4; ++mi)
      load_a_f32(&sm.a.S[h][0][0], 64, mi * 16, ks, lane, pah[mi], pal[mi]);
#pragma unroll
    for (int nt = 0; nt < 2; ++nt) {
      v16h bh, bl;
      load_b_f32(&sm.a.qkv[0][512 + h * 32], 768, ks, nt * 16, lane, bh, bl);
#pragma unroll
      for (int mi = 0; mi < 4; ++mi)
        oacc[mi][nt] = wmma3(pah[mi], pal[mi], bh, bl, oacc[mi][nt]);
    }
  }

  // Scatter to y (packed hi/lo) with reverse shift; ch = shift*256 + head*32 + d.
#pragma unroll
  for (int mi = 0; mi < 4; ++mi)
#pragma unroll
    for (int nt = 0; nt < 2; ++nt)
#pragma unroll
      for (int v = 0; v < 8; ++v) {
        const int row = mi * 16 + v + 8 * hf, d = nt * 16 + lr;
        const int hs = wy * 8 + (row >> 3), wsS = wx * 8 + (row & 7);
        const int hdst = (hs + s0) & 63, wdst = (wsS + s1) & 63;
        y[(((size_t)b * 64 + hdst) * 64 + wdst) * 1024 + shift * 256 + h * 32 + d] =
            pack_hl(oacc[mi][nt][v]);
      }
}

// ---- Kernel 3: proj GEMM  out = y @ proj_w^T + b ---------------------------
// grid = (256 M-blocks of 128 rows, 8 N-blocks of 128 cols), block = 256.
// A-tiles (128x32 packed u32 = 16 KB) staged via GLOBAL_LOAD_ASYNC_TO_LDS_B128.
#define YLD 36  // padded tile stride (u32): 36 dwords -> conflict-free A loads
__global__ __launch_bounds__(256) void proj_gemm(
    const unsigned* __restrict__ y,
    const _Float16* __restrict__ pwhi, const _Float16* __restrict__ pwlo,
    const float* __restrict__ proj_b,
    float* __restrict__ out) {
  __shared__ unsigned yt[128][YLD];  // 18.4 KB
  const int tid = threadIdx.x, lane = tid & 31, wave = tid >> 5;
  const int hf = (lane >> 4) & 1, lr = lane & 15;
  const int m2 = wave >> 2, n4 = wave & 3;  // 2x4 wave grid: 64 rows x 32 cols each
  const size_t mbase = (size_t)blockIdx.x * 128;
  const int nbase = blockIdx.y * 128 + n4 * 32;

  v8f acc[4][2] = {};
#pragma unroll 1
  for (int k0 = 0; k0 < 1024; k0 += 32) {
    __syncthreads();  // previous tile fully consumed
    // async-stage this K-slab of y: 128 rows x 32 u32, 16B chunks
    for (int i = tid; i < 128 * 8; i += 256) {
      const int r = i >> 3, c4 = (i & 7) * 4;
      const unsigned lds_off = (unsigned)(uintptr_t)&yt[r][c4];
      async_copy_b128(lds_off, y + (mbase + r) * 1024 + k0 + c4);
    }
    wait_async0();
    __syncthreads();  // all waves' async copies visible

    v16h ah[4], al[4];
#pragma unroll
    for (int mi = 0; mi < 4; ++mi)
      load_a_pk(&yt[0][0], YLD, m2 * 64 + mi * 16, 0, lane, ah[mi], al[mi]);
#pragma unroll
    for (int nt = 0; nt < 2; ++nt) {
      v16h bh, bl;
      load_b_wT_h(pwhi, pwlo, 1024, k0, nbase + nt * 16, lane, bh, bl);
#pragma unroll
      for (int mi = 0; mi < 4; ++mi)
        acc[mi][nt] = wmma3(ah[mi], al[mi], bh, bl, acc[mi][nt]);
    }
  }
#pragma unroll
  for (int nt = 0; nt < 2; ++nt) {
    const float bias = proj_b[nbase + nt * 16 + lr];
#pragma unroll
    for (int mi = 0; mi < 4; ++mi)
#pragma unroll
      for (int v = 0; v < 8; ++v)
        out[(mbase + m2 * 64 + mi * 16 + v + 8 * hf) * 1024 + nbase + nt * 16 + lr] =
            acc[mi][nt][v] + bias;
  }
}

extern "C" void kernel_launch(void* const* d_in, const int* in_sizes, int n_in,
                              void* d_out, int out_size, void* d_ws, size_t ws_size,
                              hipStream_t stream) {
  const float* x      = (const float*)d_in[0];
  const float* qkv_w  = (const float*)d_in[1];
  const float* qkv_b  = (const float*)d_in[2];
  const float* proj_w = (const float*)d_in[3];
  const float* proj_b = (const float*)d_in[4];
  float* out = (float*)d_out;

  const size_t Y_BYTES = (size_t)8 * 64 * 64 * 1024 * 4;  // 134217728
  const int QW = 3072 * 1024;
  const int PW = 1024 * 1024;
  unsigned* y = (unsigned*)d_ws;
  _Float16* qwhi = (_Float16*)((char*)d_ws + Y_BYTES);
  _Float16* qwlo = qwhi + QW;
  _Float16* pwhi = qwlo + QW;
  _Float16* pwlo = pwhi + PW;

  wsplit<<<(QW + 255) / 256, 256, 0, stream>>>(qkv_w, qwhi, qwlo, QW);
  wsplit<<<(PW + 255) / 256, 256, 0, stream>>>(proj_w, pwhi, pwlo, PW);
  attn_fused<<<dim3(512, 4), 256, 0, stream>>>(x, qwhi, qwlo, qkv_b, y);
  proj_gemm<<<dim3(256, 8), 256, 0, stream>>>(y, pwhi, pwlo, proj_b, out);
}
